// LinearTimeSelfAttention_56195352101302
// MI455X (gfx1250) — matrix operand order repaired and checked by reading, not compiled, not run on hardware
//
#include <hip/hip_runtime.h>
#include <hip/hip_bf16.h>

// ---- CDNA5 WMMA types -------------------------------------------------------
typedef __attribute__((ext_vector_type(16))) __bf16 v16bf;
typedef __attribute__((ext_vector_type(8)))  float  v8f;

#define HEADS   8
#define DHEAD   64
#define C_IN    256
#define HID     512          // HEADS*DHEAD
#define NSP     65536        // H*W
#define NGROUPS 32

union FragAB { unsigned int u[8]; v16bf v; };

__device__ __forceinline__ unsigned short f2bf(float f) {
    // round-to-nearest-even float -> bf16 (bit level)
    unsigned int u = __float_as_uint(f);
    unsigned int r = u + 0x7fffu + ((u >> 16) & 1u);
    return (unsigned short)(r >> 16);
}
__device__ __forceinline__ unsigned int packbf(float lo, float hi) {
    return (unsigned int)f2bf(lo) | ((unsigned int)f2bf(hi) << 16);
}

// A-fragment (16x32 bf16, row M = lane&15): per-VGPR K pair at
// (v>>2)*16 + half*8 + (v&3)*2 -> two contiguous 16B groups => 2x b128 loads.
__device__ __forceinline__ v16bf load_frag_a(const unsigned short* __restrict__ row,
                                             int kk, int half) {
    FragAB a;
    #pragma unroll
    for (int v = 0; v < 8; ++v)
        a.u[v] = *(const unsigned int*)(row + kk + (v >> 2) * 16 + half * 8 + (v & 3) * 2);
    return a.v;
}
// B-fragment (32x16 bf16, col N = lane&15): K pairs at
// (v>>2)*16 + half*8 + (v&3)*2 -> two contiguous 16B groups => 2x b128 loads.
__device__ __forceinline__ v16bf load_frag_b(const unsigned short* __restrict__ col,
                                             int kk, int half) {
    FragAB b;
    #pragma unroll
    for (int v = 0; v < 8; ++v)
        b.u[v] = *(const unsigned int*)(col + kk + (v >> 2) * 16 + half * 8 + (v & 3) * 2);
    return b.v;
}
#define WMMA_BF16(A, B, C) \
    __builtin_amdgcn_wmma_f32_16x16x32_bf16(false, (A), false, (B), (short)0, (C), false, false)

// ---- K0: convert weights to bf16 -------------------------------------------
__global__ __launch_bounds__(256)
void k_convert_weights(const float* __restrict__ qkv_w, const float* __restrict__ out_w,
                       unsigned short* __restrict__ qkvw_bf, unsigned short* __restrict__ outw_bf) {
    int i = blockIdx.x * 256 + threadIdx.x;
    if (i < 1536 * C_IN) qkvw_bf[i] = f2bf(qkv_w[i]);
    if (i < C_IN * HID)  outw_bf[i] = f2bf(out_w[i]);
}

// ---- K1: GroupNorm statistics (one block per group) -------------------------
__global__ __launch_bounds__(256)
void k_gn_stats(const float* __restrict__ x, float* __restrict__ mean, float* __restrict__ rstd) {
    __shared__ float s1[256], s2[256];
    int g = blockIdx.x;
    const float* p = x + (size_t)g * 8 * NSP;      // channels of a group are contiguous
    float s = 0.f, sq = 0.f;
    for (int i = threadIdx.x; i < 8 * NSP; i += 256) { float v = p[i]; s += v; sq += v * v; }
    s1[threadIdx.x] = s; s2[threadIdx.x] = sq;
    __syncthreads();
    for (int off = 128; off > 0; off >>= 1) {
        if (threadIdx.x < off) {
            s1[threadIdx.x] += s1[threadIdx.x + off];
            s2[threadIdx.x] += s2[threadIdx.x + off];
        }
        __syncthreads();
    }
    if (threadIdx.x == 0) {
        const float inv = 1.f / (8.f * NSP);
        float m = s1[0] * inv;
        float var = s2[0] * inv - m * m;
        mean[g] = m;
        rstd[g] = rsqrtf(var + 1e-5f);
    }
}

// ---- K2: apply GN, write xn transposed (n-major) as bf16, 16B stores --------
// blockIdx.x = (hwBlock<<5) | groupId ; thread handles 8 channels of one group
// at one spatial position -> coalesced f32 reads, aligned b128 bf16 store.
__global__ __launch_bounds__(256)
void k_gn_apply(const float* __restrict__ x, const float* __restrict__ gn_w,
                const float* __restrict__ gn_b, const float* __restrict__ mean,
                const float* __restrict__ rstd, unsigned short* __restrict__ xnt) {
    int g  = blockIdx.x & 31;                    // group id == c0/8 (block-uniform)
    int hw = (blockIdx.x >> 5) * 256 + threadIdx.x;
    int c0 = g * 8;
    float m = mean[g], rs = rstd[g];
    unsigned int pk[4];
    #pragma unroll
    for (int i = 0; i < 4; ++i) {
        int c = c0 + 2 * i;
        float v0 = (x[(size_t)c * NSP + hw]       - m) * rs * gn_w[c]     + gn_b[c];
        float v1 = (x[(size_t)(c + 1) * NSP + hw] - m) * rs * gn_w[c + 1] + gn_b[c + 1];
        pk[i] = packbf(v0, v1);
    }
    *(uint4*)(xnt + (size_t)hw * C_IN + c0) = make_uint4(pk[0], pk[1], pk[2], pk[3]);
}

// ---- K3: QKV GEMM (1536 x 65536 x 256), 2x4 register-blocked WMMA -----------
__global__ __launch_bounds__(256)
void k_qkv_gemm(const unsigned short* __restrict__ qkvw_bf, const float* __restrict__ qkv_b,
                const unsigned short* __restrict__ xnt,
                unsigned short* __restrict__ q_t, float* __restrict__ k_f,
                unsigned short* __restrict__ v_bf) {
    int wid  = blockIdx.x * 8 + (threadIdx.x >> 5);
    int lane = threadIdx.x & 31;
    int nt = wid & 1023;            // 1024 n-tiles of 64
    int ot = wid >> 10;             // 48 o-tiles of 32
    int o0 = ot * 32, n0 = nt * 64;
    int half = lane >> 4, l15 = lane & 15;

    const unsigned short* arow0 = qkvw_bf + (size_t)(o0 + l15) * C_IN;
    const unsigned short* arow1 = arow0 + 16 * C_IN;
    const unsigned short* bc0 = xnt + (size_t)(n0 + l15) * C_IN;
    v8f acc[2][4] = {};
    for (int kk = 0; kk < C_IN; kk += 32) {
        v16bf a0 = load_frag_a(arow0, kk, half);
        v16bf a1 = load_frag_a(arow1, kk, half);
        #pragma unroll
        for (int j = 0; j < 4; ++j) {
            v16bf b = load_frag_b(bc0 + (size_t)(j * 16) * C_IN, kk, half);
            acc[0][j] = WMMA_BF16(a0, b, acc[0][j]);
            acc[1][j] = WMMA_BF16(a1, b, acc[1][j]);
        }
    }
    #pragma unroll
    for (int i = 0; i < 2; ++i) {
        #pragma unroll
        for (int j = 0; j < 4; ++j) {
            int n = n0 + j * 16 + l15;
            #pragma unroll
            for (int r = 0; r < 8; ++r) {
                int o = o0 + i * 16 + r + 8 * half;
                float val = acc[i][j][r] + qkv_b[o];
                if (o < HID) {                                    // q: transposed bf16
                    q_t[(size_t)n * HID + o] = f2bf(val);
                } else if (o < 2 * HID) {                         // k: f32 (softmax)
                    k_f[(size_t)(o - HID) * NSP + n] = val;
                } else {                                          // v: bf16 row-major
                    v_bf[(size_t)(o - 2 * HID) * NSP + n] = f2bf(val);
                }
            }
        }
    }
}

// ---- K4: softmax row statistics over spatial axis (one block per row) -------
__global__ __launch_bounds__(256)
void k_softmax_stats(const float* __restrict__ k_f,
                     float* __restrict__ rowmax, float* __restrict__ rowsum) {
    __shared__ float sm[256];
    int row = blockIdx.x;
    const float* p = k_f + (size_t)row * NSP;
    float m = -3.4e38f;
    for (int i = threadIdx.x; i < NSP; i += 256) m = fmaxf(m, p[i]);
    sm[threadIdx.x] = m; __syncthreads();
    for (int off = 128; off > 0; off >>= 1) {
        if (threadIdx.x < off) sm[threadIdx.x] = fmaxf(sm[threadIdx.x], sm[threadIdx.x + off]);
        __syncthreads();
    }
    m = sm[0];
    __syncthreads();
    float s = 0.f;
    for (int i = threadIdx.x; i < NSP; i += 256) s += expf(p[i] - m);
    sm[threadIdx.x] = s; __syncthreads();
    for (int off = 128; off > 0; off >>= 1) {
        if (threadIdx.x < off) sm[threadIdx.x] += sm[threadIdx.x + off];
        __syncthreads();
    }
    if (threadIdx.x == 0) { rowmax[row] = m; rowsum[row] = sm[0]; }
}

__global__ __launch_bounds__(256)
void k_zero_f32(float* __restrict__ p, int n) {
    int i = blockIdx.x * 256 + threadIdx.x;
    if (i < n) p[i] = 0.f;
}

// ---- K5: context = exp(k - max) @ v^T, 1x4 blocked: exp-fragment reused -----
__global__ __launch_bounds__(256)
void k_context_gemm(const float* __restrict__ k_f, const float* __restrict__ rowmax,
                    const unsigned short* __restrict__ v_bf, float* __restrict__ ctx) {
    int wid  = blockIdx.x * 8 + (threadIdx.x >> 5);
    int lane = threadIdx.x & 31;
    int slice = wid & 127;             // 128 slices of n (512 each)
    int t = wid >> 7;                  // 0..31 = h(8) x dt(4)
    int h = t >> 2, dt = t & 3;
    int half = lane >> 4, l15 = lane & 15;

    int drow = h * DHEAD + dt * 16 + l15;
    float rm = rowmax[drow];
    const float* kp          = k_f  + (size_t)drow * NSP;
    const unsigned short* vp = v_bf + (size_t)(h * DHEAD + l15) * NSP;

    v8f acc[4] = {};
    int nbase = slice * 512;
    for (int c = 0; c < 512; c += 32) {
        int nn = nbase + c;
        FragAB a;
        #pragma unroll
        for (int v = 0; v < 8; ++v) {
            int ka = nn + (v >> 2) * 16 + half * 8 + (v & 3) * 2;
            a.u[v] = packbf(expf(kp[ka] - rm), expf(kp[ka + 1] - rm));
        }
        #pragma unroll
        for (int et = 0; et < 4; ++et) {
            v16bf b = load_frag_b(vp + (size_t)(et * 16) * NSP, nn, half);
            acc[et] = WMMA_BF16(a.v, b, acc[et]);
        }
    }
    #pragma unroll
    for (int et = 0; et < 4; ++et) {
        #pragma unroll
        for (int r = 0; r < 8; ++r) {
            int d = dt * 16 + r + 8 * half;
            int e = et * 16 + l15;
            atomicAdd(&ctx[(size_t)(h * DHEAD + d) * DHEAD + e], acc[et][r]);
        }
    }
}

// ---- K6: normalize context by softmax denom, store transposed bf16 ----------
__global__ __launch_bounds__(256)
void k_ctx_norm(const float* __restrict__ ctx, const float* __restrict__ rowsum,
                unsigned short* __restrict__ ct) {
    int i = blockIdx.x * 256 + threadIdx.x;   // 8*64*64 = 32768
    int row = i >> 6;                         // h*64 + d
    int e = i & 63;
    int h = row >> 6, d = row & 63;
    float v = ctx[i] / rowsum[row];
    ct[(size_t)h * 4096 + e * 64 + d] = f2bf(v);   // ct[h][e][d]: A row-major for K7
}

// ---- K7: attn = context^T @ q per head, 2x4 blocked; write transposed bf16 --
__global__ __launch_bounds__(256)
void k_attn_gemm(const unsigned short* __restrict__ ct, const unsigned short* __restrict__ q_t,
                 unsigned short* __restrict__ attn_t) {
    int wid  = blockIdx.x * 8 + (threadIdx.x >> 5);
    int lane = threadIdx.x & 31;
    int nt = wid & 1023;               // 1024 n-tiles of 64
    int t  = wid >> 10;                // 0..15 = h(8) x et2(2)
    int et2 = t & 1, h = t >> 1;
    int half = lane >> 4, l15 = lane & 15;
    int n0 = nt * 64;

    const unsigned short* arow0 = ct + (size_t)h * 4096 + (size_t)(et2 * 32 + l15) * 64;
    const unsigned short* arow1 = arow0 + 16 * 64;
    const unsigned short* bc0 = q_t + (size_t)(n0 + l15) * HID + h * DHEAD;
    v8f acc[2][4] = {};
    #pragma unroll
    for (int kk = 0; kk < DHEAD; kk += 32) {
        v16bf a0 = load_frag_a(arow0, kk, half);
        v16bf a1 = load_frag_a(arow1, kk, half);
        #pragma unroll
        for (int j = 0; j < 4; ++j) {
            v16bf b = load_frag_b(bc0 + (size_t)(j * 16) * HID, kk, half);
            acc[0][j] = WMMA_BF16(a0, b, acc[0][j]);
            acc[1][j] = WMMA_BF16(a1, b, acc[1][j]);
        }
    }
    #pragma unroll
    for (int i = 0; i < 2; ++i) {
        #pragma unroll
        for (int j = 0; j < 4; ++j) {
            int n = n0 + j * 16 + l15;
            #pragma unroll
            for (int r = 0; r < 8; ++r) {
                int ch = h * DHEAD + et2 * 32 + i * 16 + r + 8 * half;
                attn_t[(size_t)n * HID + ch] = f2bf(acc[i][j][r]);
            }
        }
    }
}

// ---- K8: final projection (256 x 65536 x 512), 2x4 blocked + bias -----------
__global__ __launch_bounds__(256)
void k_out_gemm(const unsigned short* __restrict__ outw_bf,
                const unsigned short* __restrict__ attn_t,
                const float* __restrict__ out_b, float* __restrict__ y) {
    int wid  = blockIdx.x * 8 + (threadIdx.x >> 5);
    int lane = threadIdx.x & 31;
    int nt = wid & 1023, ot = wid >> 10;     // 8 o-tiles(32) x 1024 n-tiles(64)
    int half = lane >> 4, l15 = lane & 15;
    int o0 = ot * 32, n0 = nt * 64;

    const unsigned short* arow0 = outw_bf + (size_t)(o0 + l15) * HID;
    const unsigned short* arow1 = arow0 + 16 * HID;
    const unsigned short* bc0 = attn_t + (size_t)(n0 + l15) * HID;
    v8f acc[2][4] = {};
    for (int kk = 0; kk < HID; kk += 32) {
        v16bf a0 = load_frag_a(arow0, kk, half);
        v16bf a1 = load_frag_a(arow1, kk, half);
        #pragma unroll
        for (int j = 0; j < 4; ++j) {
            v16bf b = load_frag_b(bc0 + (size_t)(j * 16) * HID, kk, half);
            acc[0][j] = WMMA_BF16(a0, b, acc[0][j]);
            acc[1][j] = WMMA_BF16(a1, b, acc[1][j]);
        }
    }
    #pragma unroll
    for (int i = 0; i < 2; ++i) {
        #pragma unroll
        for (int j = 0; j < 4; ++j) {
            int n = n0 + j * 16 + l15;
            #pragma unroll
            for (int r = 0; r < 8; ++r) {
                int o = o0 + i * 16 + r + 8 * half;
                y[(size_t)o * NSP + n] = acc[i][j][r] + out_b[o];
            }
        }
    }
}

// ---- workspace layout (bytes) ----------------------------------------------
#define WS_MEAN    (size_t)0                      // 32 f32
#define WS_RSTD    (size_t)128                    // 32 f32
#define WS_QKVW    (size_t)256                    // 1536*256 bf16 = 786432
#define WS_OUTW    (size_t)786688                 // 256*512 bf16  = 262144
#define WS_ROWMAX  (size_t)1048832                // 512 f32
#define WS_ROWSUM  (size_t)1050880                // 512 f32
#define WS_CTX     (size_t)1052928                // 8*64*64 f32 = 131072
#define WS_CTXBF   (size_t)1184000                // 8*64*64 bf16 = 65536
#define WS_XNT     (size_t)2097152                // 65536*256 bf16 = 32 MiB
#define WS_QT      (size_t)35651584               // 65536*512 bf16 = 64 MiB
#define WS_KF      (size_t)102760448              // 512*65536 f32 = 128 MiB
#define WS_VBF     (size_t)236978176              // 512*65536 bf16 = 64 MiB
#define WS_ATTNT   (size_t)304087040              // 65536*512 bf16 = 64 MiB
// total ~354 MiB

extern "C" void kernel_launch(void* const* d_in, const int* in_sizes, int n_in,
                              void* d_out, int out_size, void* d_ws, size_t ws_size,
                              hipStream_t stream) {
    const float* x     = (const float*)d_in[0];
    const float* gn_w  = (const float*)d_in[1];
    const float* gn_b  = (const float*)d_in[2];
    const float* qkv_w = (const float*)d_in[3];
    const float* qkv_b = (const float*)d_in[4];
    const float* out_w = (const float*)d_in[5];
    const float* out_b = (const float*)d_in[6];
    float* y = (float*)d_out;

    char* ws = (char*)d_ws;
    float*          mean    = (float*)(ws + WS_MEAN);
    float*          rstd    = (float*)(ws + WS_RSTD);
    unsigned short* qkvw_bf = (unsigned short*)(ws + WS_QKVW);
    unsigned short* outw_bf = (unsigned short*)(ws + WS_OUTW);
    float*          rowmax  = (float*)(ws + WS_ROWMAX);
    float*          rowsum  = (float*)(ws + WS_ROWSUM);
    float*          ctx     = (float*)(ws + WS_CTX);
    unsigned short* ctxbf   = (unsigned short*)(ws + WS_CTXBF);
    unsigned short* xnt     = (unsigned short*)(ws + WS_XNT);
    unsigned short* q_t     = (unsigned short*)(ws + WS_QT);
    float*          k_f     = (float*)(ws + WS_KF);
    unsigned short* v_bf    = (unsigned short*)(ws + WS_VBF);
    unsigned short* attn_t  = (unsigned short*)(ws + WS_ATTNT);

    k_convert_weights<<<1536, 256, 0, stream>>>(qkv_w, out_w, qkvw_bf, outw_bf);
    k_gn_stats<<<NGROUPS, 256, 0, stream>>>(x, mean, rstd);
    k_gn_apply<<<(NSP / 256) * 32, 256, 0, stream>>>(x, gn_w, gn_b, mean, rstd, xnt);
    k_qkv_gemm<<<(48 * 1024) / 8, 256, 0, stream>>>(qkvw_bf, qkv_b, xnt, q_t, k_f, v_bf);
    k_softmax_stats<<<HID, 256, 0, stream>>>(k_f, rowmax, rowsum);
    k_zero_f32<<<(HEADS * DHEAD * DHEAD) / 256, 256, 0, stream>>>(ctx, HEADS * DHEAD * DHEAD);
    k_context_gemm<<<(32 * 128) / 8, 256, 0, stream>>>(k_f, rowmax, v_bf, ctx);
    k_ctx_norm<<<(HEADS * DHEAD * DHEAD) / 256, 256, 0, stream>>>(ctx, rowsum, ctxbf);
    k_attn_gemm<<<(16 * 1024) / 8, 256, 0, stream>>>(ctxbf, q_t, attn_t);
    k_out_gemm<<<(8 * 1024) / 8, 256, 0, stream>>>(outw_bf, attn_t, out_b, y);
}
